// VQVAE_14027363189270
// MI455X (gfx1250) — matrix operand order repaired and checked by reading, not compiled, not run on hardware
//
#include <hip/hip_runtime.h>
#include <hip/hip_bf16.h>

// ---------------------------------------------------------------------------
// Types for CDNA5 WMMA (wave32): V_WMMA_F32_16X16X32_BF16
// ---------------------------------------------------------------------------
typedef __attribute__((ext_vector_type(16))) __bf16 v16bf;
typedef __attribute__((ext_vector_type(8)))  float  v8f;

union V16BF { v16bf v; unsigned short us[16]; };

__device__ __forceinline__ unsigned short f32_to_bf16_bits(float f) {
    union { float f; unsigned int u; } c; c.f = f;
    unsigned int u = c.u;
    u += 0x7FFFu + ((u >> 16) & 1u);        // round-to-nearest-even
    return (unsigned short)(u >> 16);
}

__device__ __forceinline__ unsigned int float_to_ordered(float f) {
    union { float f; unsigned int u; } c; c.f = f;
    unsigned int u = c.u;
    return (u & 0x80000000u) ? ~u : (u | 0x80000000u);   // monotone ascending
}

#define ACT_NONE  0
#define ACT_RELU  1
#define ACT_PRELU 2

#define BM  64     // out-channel macro-tile
#define BN  128    // pixel macro-tile
#define BK  32     // K chunk
#define BKP 40     // padded LDS row (ushorts): 80B = 20 banks -> conflict-free b128

// ---------------------------------------------------------------------------
// LDS-staged, software-pipelined implicit-GEMM convolution.
// Block = 256 threads (8 waves) computes a BM x BN macro-tile of
// GEMM( A = weights (OC x K), B = im2col (K x N) ), K = IC*KH*KW.
// Wave w -> M-subtile (w&3), N-subtiles 4*(w>>2)..+3 : 4 WMMAs per chunk
// off a single A fragment. Next chunk's global loads overlap the WMMAs.
// ---------------------------------------------------------------------------
__global__ __launch_bounds__(256)
void conv_wmma_kernel(const float* __restrict__ X, const float* __restrict__ Wt,
                      const float* __restrict__ bias, const float* __restrict__ slope,
                      float* __restrict__ Y,
                      int Bn, int IC, int IH, int IW,
                      int OC, int OH, int OW,
                      int KH, int KW, int stride, int pad, int act)
{
    __shared__ unsigned short As[BM][BKP];   // [m][k], k contiguous
    __shared__ unsigned short Bs[BN][BKP];   // [n][k], k contiguous

    const int tid  = threadIdx.x;
    const int lane = tid & 31;
    const int wave = tid >> 5;
    const int l16  = lane & 15;
    const int half = lane >> 4;

    const int NT   = Bn * OH * OW;
    const int Ntb  = (NT + BN - 1) / BN;
    const int mtb  = blockIdx.x / Ntb;
    const int ntb  = blockIdx.x % Ntb;

    const int K   = IC * KH * KW;
    const int KHW = KH * KW;

    // ---- cooperative-fill assignment (fixed per thread) ----
    // A tile: 64 rows x 32 k = 2048 el; thread t -> row t>>2, k run (t&3)*8
    const int fmA  = tid >> 2;
    const int fkA  = (tid & 3) * 8;
    const int gm   = mtb * BM + fmA;
    const bool gmv = (gm < OC);
    // B tile: 128 cols x 32 k = 4096 el; thread t -> col t>>1, k run (t&1)*16
    const int fnB  = tid >> 1;
    const int fkB  = (tid & 1) * 16;
    const int gn   = ntb * BN + fnB;
    const bool gnv = (gn < NT);
    int bbf = 0, ohf = 0, owf = 0;
    if (gnv) { bbf = gn / (OH * OW); int r = gn % (OH * OW); ohf = r / OW; owf = r % OW; }
    const int ih0f = ohf * stride - pad;
    const int iw0f = owf * stride - pad;

    // ---- wave sub-tile assignment ----
    const int wm  = wave & 3;              // M subtile 0..3
    const int wn0 = (wave >> 2) * 4;       // N subtiles wn0 .. wn0+3
    const int am  = wm * 16 + l16;         // A fragment row in As

    v8f acc[4];
    #pragma unroll
    for (int t = 0; t < 4; ++t) acc[t] = (v8f){0.f,0.f,0.f,0.f,0.f,0.f,0.f,0.f};

    // ---- prefetch registers ----
    float ra[8], rb[16];

    auto loadA = [&](int k0) {
        #pragma unroll
        for (int j = 0; j < 8; ++j) {
            const int k = k0 + fkA + j;
            ra[j] = (gmv && k < K) ? Wt[(long long)gm * K + k] : 0.f;
        }
    };
    auto loadB = [&](int k0) {
        #pragma unroll
        for (int j = 0; j < 16; ++j) {
            const int k = k0 + fkB + j;
            float v = 0.f;
            if (gnv && k < K) {
                const int ic = k / KHW;
                const int rr = k - ic * KHW;
                const int r  = rr / KW;
                const int s2 = rr - r * KW;
                const int ih = ih0f + r, iw = iw0f + s2;
                if ((unsigned)ih < (unsigned)IH && (unsigned)iw < (unsigned)IW)
                    v = X[(((long long)bbf * IC + ic) * IH + ih) * IW + iw];
            }
            rb[j] = v;
        }
    };

    loadA(0);
    loadB(0);

    for (int k0 = 0; k0 < K; k0 += BK) {
        // ---- convert + store staged chunk to LDS ----
        {
            union { unsigned short us[8]; uint4 q; } ta;
            #pragma unroll
            for (int j = 0; j < 8; ++j) ta.us[j] = f32_to_bf16_bits(ra[j]);
            *(uint4*)&As[fmA][fkA] = ta.q;

            union { unsigned short us[16]; uint4 q[2]; } tb;
            #pragma unroll
            for (int j = 0; j < 16; ++j) tb.us[j] = f32_to_bf16_bits(rb[j]);
            *(uint4*)&Bs[fnB][fkB]     = tb.q[0];
            *(uint4*)&Bs[fnB][fkB + 8] = tb.q[1];
        }
        __syncthreads();

        // ---- prefetch next chunk (overlaps ds loads + WMMAs below) ----
        if (k0 + BK < K) { loadA(k0 + BK); loadB(k0 + BK); }

        // ---- fragments from LDS (aligned 128-bit runs) + 4 WMMAs ----
        union { v16bf v; uint4 q[2]; } a;
        a.q[0] = *(const uint4*)&As[am][8 * half];        // k = 8h .. 8h+7
        a.q[1] = *(const uint4*)&As[am][16 + 8 * half];   // k = 16+8h .. +7
        #pragma unroll
        for (int t = 0; t < 4; ++t) {
            const int bn = (wn0 + t) * 16 + l16;
            union { v16bf v; uint4 q[2]; } b;
            b.q[0] = *(const uint4*)&Bs[bn][8 * half];        // k = 8h .. 8h+7
            b.q[1] = *(const uint4*)&Bs[bn][16 + 8 * half];   // k = 16+8h .. +7
            acc[t] = __builtin_amdgcn_wmma_f32_16x16x32_bf16(false, a.v, false, b.v,
                                                             (short)0, acc[t], false, false);
        }
        __syncthreads();
    }

    // ---- epilogue: bias + activation, C layout M = v + 8*half, N = l16 ----
    #pragma unroll
    for (int t = 0; t < 4; ++t) {
        const int nn = ntb * BN + (wn0 + t) * 16 + l16;
        if (nn < NT) {
            const int bb = nn / (OH * OW);
            int r = nn % (OH * OW);
            const int oh = r / OW, ow = r % OW;
            #pragma unroll
            for (int v = 0; v < 8; ++v) {
                const int oc = mtb * BM + wm * 16 + v + 8 * half;
                if (oc < OC) {
                    float y = acc[t][v] + bias[oc];
                    if (act == ACT_RELU)       y = fmaxf(y, 0.f);
                    else if (act == ACT_PRELU) { const float a0 = slope[0]; y = (y >= 0.f) ? y : a0 * y; }
                    Y[(((long long)bb * OC + oc) * OH + oh) * OW + ow] = y;
                }
            }
        }
    }
}

// ---------------------------------------------------------------------------
// Training-mode BatchNorm statistics: one block per channel, biased variance.
// ---------------------------------------------------------------------------
__global__ __launch_bounds__(256)
void bn_stats_kernel(const float* __restrict__ X, float* __restrict__ mean,
                     float* __restrict__ var, int C, int Bn, int HW)
{
    const int c   = blockIdx.x;
    const int tid = threadIdx.x;
    const long long per = (long long)Bn * HW;
    float s = 0.f, ss = 0.f;
    for (long long i = tid; i < per; i += blockDim.x) {
        const int b = (int)(i / HW);
        const int p = (int)(i - (long long)b * HW);
        const float v = X[((long long)b * C + c) * HW + p];
        s += v; ss += v * v;
    }
    __shared__ float sh[256], sh2[256];
    sh[tid] = s; sh2[tid] = ss; __syncthreads();
    for (int o = 128; o > 0; o >>= 1) {
        if (tid < o) { sh[tid] += sh[tid + o]; sh2[tid] += sh2[tid + o]; }
        __syncthreads();
    }
    if (tid == 0) {
        const float m = sh[0] / (float)per;
        mean[c] = m;
        var[c]  = sh2[0] / (float)per - m * m;
    }
}

// BN (+ optional scalar PReLU) in place.
__global__ __launch_bounds__(256)
void bn_prelu_kernel(float* __restrict__ X, const float* __restrict__ mean,
                     const float* __restrict__ var, const float* __restrict__ g,
                     const float* __restrict__ b, const float* __restrict__ slope,
                     float eps, int C, int HW, long long total)
{
    long long i = (long long)blockIdx.x * blockDim.x + threadIdx.x;
    if (i >= total) return;
    const int c = (int)((i / HW) % C);
    float y = g[c] * (X[i] - mean[c]) * rsqrtf(var[c] + eps) + b[c];
    if (slope) { const float a = slope[0]; y = (y >= 0.f) ? y : a * y; }
    X[i] = y;
}

// Residual tail: H += BN(T)
__global__ __launch_bounds__(256)
void bn_add_kernel(float* __restrict__ H, const float* __restrict__ T,
                   const float* __restrict__ mean, const float* __restrict__ var,
                   const float* __restrict__ g, const float* __restrict__ b,
                   float eps, int C, int HW, long long total)
{
    long long i = (long long)blockIdx.x * blockDim.x + threadIdx.x;
    if (i >= total) return;
    const int c = (int)((i / HW) % C);
    H[i] += g[c] * (T[i] - mean[c]) * rsqrtf(var[c] + eps) + b[c];
}

// BN + pixel-shuffle(r=2) + scalar PReLU fused.
// X: (Bn, 4*C2, H, W) -> Y: (Bn, C2, 2H, 2W); channel cin = c2*4 + r1*2 + r2.
__global__ __launch_bounds__(256)
void bn_shuffle_prelu_kernel(const float* __restrict__ X, float* __restrict__ Y,
                             const float* __restrict__ mean, const float* __restrict__ var,
                             const float* __restrict__ g, const float* __restrict__ b,
                             const float* __restrict__ slope, float eps,
                             int Bn, int C2, int H, int W)
{
    const long long total = (long long)Bn * C2 * 4 * H * W;
    long long i = (long long)blockIdx.x * blockDim.x + threadIdx.x;
    if (i >= total) return;
    const int OW = 2 * W, OH = 2 * H;
    long long t = i;
    const int ow = (int)(t % OW); t /= OW;
    const int oh = (int)(t % OH); t /= OH;
    const int c2 = (int)(t % C2); t /= C2;
    const int bb = (int)t;
    const int w = ow >> 1, r2 = ow & 1;
    const int h = oh >> 1, r1 = oh & 1;
    const int cin = c2 * 4 + r1 * 2 + r2;
    const float xv = X[(((long long)bb * (4 * C2) + cin) * H + h) * W + w];
    float y = g[cin] * (xv - mean[cin]) * rsqrtf(var[cin] + eps) + b[cin];
    const float a = slope[0];
    Y[i] = (y >= 0.f) ? y : a * y;
}

__global__ __launch_bounds__(256)
void add_inplace_kernel(float* __restrict__ A, const float* __restrict__ B, long long n)
{
    long long i = (long long)blockIdx.x * blockDim.x + threadIdx.x;
    if (i < n) A[i] += B[i];
}

// ---------------------------------------------------------------------------
// Vector quantizer
// ---------------------------------------------------------------------------
__global__ __launch_bounds__(256)
void enorm_kernel(const float* __restrict__ E, float* __restrict__ en, int KE, int D)
{
    const int k = blockIdx.x * blockDim.x + threadIdx.x;
    if (k < KE) {
        float s = 0.f;
        for (int d = 0; d < D; ++d) { const float v = E[(long long)k * D + d]; s += v * v; }
        en[k] = s;
    }
}

__global__ __launch_bounds__(256)
void vq_reset_kernel(unsigned long long* __restrict__ best, int n,
                     int* __restrict__ hist, int kh, float* __restrict__ loss)
{
    const int i = blockIdx.x * blockDim.x + threadIdx.x;
    if (i < n)  best[i] = ~0ull;
    if (i < kh) hist[i] = 0;
    if (i == 0) loss[0] = 0.f;
}

// Argmin_k ( ||e_k||^2 - 2 z_n . e_k ) via WMMA dot tiles.
__global__ __launch_bounds__(256)
void vq_dist_wmma_kernel(const float* __restrict__ Z, const float* __restrict__ E,
                         const float* __restrict__ en,
                         unsigned long long* __restrict__ best,
                         int NPTS, int KE, int D, int HW)
{
    const int lane = threadIdx.x & 31;
    const int wave = threadIdx.x >> 5;
    const int l16  = lane & 15;
    const int half = lane >> 4;

    const int Et = KE >> 4;
    long long tile = (long long)blockIdx.x * (blockDim.x >> 5) + wave;
    if (tile >= (long long)(NPTS >> 4) * Et) return;
    const int pt = (int)(tile / Et);
    const int et = (int)(tile % Et);

    const int m  = pt * 16 + l16;           // point for A fragment
    const int bb = m / HW;
    const int hw = m - bb * HW;
    const int e  = et * 16 + l16;           // entry for B fragment

    v8f acc = {0.f, 0.f, 0.f, 0.f, 0.f, 0.f, 0.f, 0.f};
    for (int k0 = 0; k0 < D; k0 += 32) {
        V16BF a, b;
        #pragma unroll
        for (int j = 0; j < 16; ++j) {
            const int g = j >> 1, s = j & 1;
            const int ka = k0 + ((g < 4) ? (2*g + s) : (16 + 2*(g - 4) + s)) + 8*half;
            a.us[j] = f32_to_bf16_bits(Z[((long long)bb * D + ka) * HW + hw]);
            const int kb = k0 + ((g < 4) ? (2*g + s) : (16 + 2*(g - 4) + s)) + 8*half;
            b.us[j] = f32_to_bf16_bits(E[(long long)e * D + kb]);
        }
        acc = __builtin_amdgcn_wmma_f32_16x16x32_bf16(false, a.v, false, b.v,
                                                      (short)0, acc, false, false);
    }

    #pragma unroll
    for (int v = 0; v < 8; ++v) {
        const int p   = pt * 16 + v + 8*half;
        const int ent = et * 16 + l16;
        const float s = en[ent] - 2.0f * acc[v];
        unsigned int hi = float_to_ordered(s);
        unsigned int lo = (unsigned int)ent;
        #pragma unroll
        for (int msk = 8; msk >= 1; msk >>= 1) {
            const unsigned int ohi = __shfl_xor(hi, msk, 32);
            const unsigned int olo = __shfl_xor(lo, msk, 32);
            if (ohi < hi || (ohi == hi && olo < lo)) { hi = ohi; lo = olo; }
        }
        if (l16 == 0 && p < NPTS)
            atomicMin(&best[p], ((unsigned long long)hi << 32) | (unsigned long long)lo);
    }
}

// Gather codebook vectors, accumulate SSE loss and histogram.
__global__ __launch_bounds__(256)
void vq_assign_kernel(const float* __restrict__ Z, const float* __restrict__ E,
                      const unsigned long long* __restrict__ best,
                      float* __restrict__ Q, float* __restrict__ loss_acc,
                      int* __restrict__ hist, int NPTS, int D, int HW)
{
    const int n = blockIdx.x * blockDim.x + threadIdx.x;
    if (n >= NPTS) return;
    const int ent = (int)(best[n] & 0xFFFFFFFFull);
    const int bb  = n / HW;
    const int hw  = n - bb * HW;
    float sq = 0.f;
    for (int d = 0; d < D; ++d) {
        const float ev = E[(long long)ent * D + d];
        const float zv = Z[((long long)bb * D + d) * HW + hw];
        Q[((long long)bb * D + d) * HW + hw] = ev;
        const float df = ev - zv;
        sq += df * df;
    }
    atomicAdd(loss_acc, sq);
    atomicAdd(&hist[ent], 1);
}

// loss = (1+CC)*MSE ; perplexity = exp(-sum p log(p+1e-10))
__global__ __launch_bounds__(256)
void vq_finalize_kernel(const float* __restrict__ loss_acc, const int* __restrict__ hist,
                        float* __restrict__ out_loss, float* __restrict__ out_perp,
                        int NPTS, int KE, long long ND, float cc)
{
    __shared__ float sh[256];
    const int tid = threadIdx.x;
    float acc = 0.f;
    for (int k = tid; k < KE; k += blockDim.x) {
        const float p = (float)hist[k] / (float)NPTS;
        acc += p * logf(p + 1e-10f);
    }
    sh[tid] = acc; __syncthreads();
    for (int o = 128; o > 0; o >>= 1) { if (tid < o) sh[tid] += sh[tid + o]; __syncthreads(); }
    if (tid == 0) {
        *out_loss = (1.f + cc) * loss_acc[0] / (float)ND;
        *out_perp = expf(-sh[0]);
    }
}

// ---------------------------------------------------------------------------
// Host side
// ---------------------------------------------------------------------------
extern "C" void kernel_launch(void* const* d_in, const int* in_sizes, int n_in,
                              void* d_out, int out_size, void* d_ws, size_t ws_size,
                              hipStream_t stream)
{
    (void)in_sizes; (void)n_in; (void)ws_size;

    // ---- Inputs (insertion-order recursive flatten of setup_inputs()) ----
    int ii = 0;
    auto nx = [&]() { return (const float*)d_in[ii++]; };
    const float* x = nx();                                           // (8,3,256,256)
    struct ConvP { const float *w, *b; };
    struct ResP  { const float *c1w,*c1b,*g1,*b1,*a1,*c2w,*c2b,*g2,*b2; };
    auto conv_p = [&]() { ConvP p; p.w = nx(); p.b = nx(); return p; };
    auto res_p  = [&]() {
        ResP r;
        r.c1w = nx(); r.c1b = nx(); r.g1 = nx(); r.b1 = nx(); r.a1 = nx();
        r.c2w = nx(); r.c2b = nx(); r.g2 = nx(); r.b2 = nx();
        return r;
    };
    ConvP enc_c1 = conv_p();                // 64 <- 3, k2 s2
    ConvP enc_c2 = conv_p();                // 128 <- 64, k2 s2
    ConvP enc_c3 = conv_p();                // 128 <- 128, k3 p1
    ResP  enc_res[2] = { res_p(), res_p() };
    ConvP prevq  = conv_p();                // 64 <- 128, 1x1
    const float* emb = nx();                // (512,64)
    ConvP dec_c1 = conv_p();                // 128 <- 64, k3 p1
    ResP  dec_res[2] = { res_p(), res_p() };
    struct UpP { const float *w,*b,*g,*bb,*a; } up[2];
    for (int u = 0; u < 2; ++u) { up[u].w = nx(); up[u].b = nx(); up[u].g = nx(); up[u].bb = nx(); up[u].a = nx(); }
    ConvP dec_c2 = conv_p();                // 64 <- 128, k3 p1
    const float* a_out = nx();
    ConvP dec_c3 = conv_p();                // 3 <- 64, k3 p1

    float* out = (float*)d_out;

    // ---- Workspace bump allocator ----
    char* ws = (char*)d_ws;
    size_t off = 0;
    auto alloc = [&](size_t bytes) -> void* {
        void* p = ws + off;
        off = (off + bytes + 255) & ~(size_t)255;
        return p;
    };
    float* G1 = (float*)alloc((size_t)8388608  * 4);   // (8,64,128,128)
    float* G2 = (float*)alloc((size_t)4194304  * 4);   // (8,128,64,64)
    float* G3 = (float*)alloc((size_t)4194304  * 4);
    float* T  = (float*)alloc((size_t)4194304  * 4);
    float* U  = (float*)alloc((size_t)4194304  * 4);
    float* X2 = (float*)alloc((size_t)4194304  * 4);
    float* Zb = (float*)alloc((size_t)2097152  * 4);   // (8,64,64,64)
    float* Qb = (float*)alloc((size_t)2097152  * 4);
    float* Ab = (float*)alloc((size_t)16777216 * 4);   // (8,512,64,64)
    float* Bb = (float*)alloc((size_t)16777216 * 4);   // (8,128,128,128)
    float* Cb = (float*)alloc((size_t)67108864 * 4);   // (8,512,128,128)
    float* Db = (float*)alloc((size_t)67108864 * 4);   // (8,128,256,256)
    unsigned long long* best = (unsigned long long*)alloc((size_t)32768 * 8);
    float* d_mean  = (float*)alloc(512 * 4);
    float* d_var   = (float*)alloc(512 * 4);
    float* d_enorm = (float*)alloc(512 * 4);
    int*   d_hist  = (int*)alloc(512 * 4);
    float* d_loss  = (float*)alloc(4);

    auto eblocks = [](long long n) { return (int)((n + 255) / 256); };

    auto conv = [&](const float* Xp, const float* Wp, const float* bp, float* Yp,
                    int Bn, int IC, int IH, int IW, int OC, int KH, int KW,
                    int st, int pad, int act, const float* slope) {
        const int OH = (IH + 2 * pad - KH) / st + 1;
        const int OW = (IW + 2 * pad - KW) / st + 1;
        const int Mt = (OC + BM - 1) / BM;
        const int Nt = (Bn * OH * OW + BN - 1) / BN;
        conv_wmma_kernel<<<Mt * Nt, 256, 0, stream>>>(Xp, Wp, bp, slope, Yp,
                                                      Bn, IC, IH, IW, OC, OH, OW,
                                                      KH, KW, st, pad, act);
    };

    auto resblock = [&](float* Hbuf, const ResP& r, float eps, int C, int Bn, int IH, int IW) {
        const int HW = IH * IW;
        const long long total = (long long)Bn * C * HW;
        conv(Hbuf, r.c1w, r.c1b, T, Bn, C, IH, IW, C, 3, 3, 1, 1, ACT_NONE, nullptr);
        bn_stats_kernel<<<C, 256, 0, stream>>>(T, d_mean, d_var, C, Bn, HW);
        bn_prelu_kernel<<<eblocks(total), 256, 0, stream>>>(T, d_mean, d_var, r.g1, r.b1,
                                                            r.a1, eps, C, HW, total);
        conv(T, r.c2w, r.c2b, U, Bn, C, IH, IW, C, 3, 3, 1, 1, ACT_NONE, nullptr);
        bn_stats_kernel<<<C, 256, 0, stream>>>(U, d_mean, d_var, C, Bn, HW);
        bn_add_kernel<<<eblocks(total), 256, 0, stream>>>(Hbuf, U, d_mean, d_var, r.g2, r.b2,
                                                          eps, C, HW, total);
    };

    const float EPS_RES = 0.8f, EPS_UP = 1e-5f, CC = 0.25f;

    // ---------------- Encoder ----------------
    conv(x,  enc_c1.w, enc_c1.b, G1, 8,   3, 256, 256,  64, 2, 2, 2, 0, ACT_RELU, nullptr);
    conv(G1, enc_c2.w, enc_c2.b, G2, 8,  64, 128, 128, 128, 2, 2, 2, 0, ACT_RELU, nullptr);
    conv(G2, enc_c3.w, enc_c3.b, G3, 8, 128,  64,  64, 128, 3, 3, 1, 1, ACT_RELU, nullptr);
    for (int r = 0; r < 2; ++r) resblock(G3, enc_res[r], EPS_RES, 128, 8, 64, 64);
    conv(G3, prevq.w, prevq.b, Zb, 8, 128, 64, 64, 64, 1, 1, 1, 0, ACT_NONE, nullptr);

    // ---------------- Vector quantizer ----------------
    const int NPTS = 32768, KE = 512, D = 64, HW = 4096;
    enorm_kernel<<<2, 256, 0, stream>>>(emb, d_enorm, KE, D);
    vq_reset_kernel<<<eblocks(NPTS), 256, 0, stream>>>(best, NPTS, d_hist, KE, d_loss);
    {
        const long long tiles = (long long)(NPTS / 16) * (KE / 16);   // 65536 tiles
        vq_dist_wmma_kernel<<<(int)((tiles + 7) / 8), 256, 0, stream>>>(Zb, emb, d_enorm,
                                                                        best, NPTS, KE, D, HW);
    }
    vq_assign_kernel<<<eblocks(NPTS), 256, 0, stream>>>(Zb, emb, best, Qb, d_loss, d_hist,
                                                        NPTS, D, HW);
    vq_finalize_kernel<<<1, 256, 0, stream>>>(d_loss, d_hist, out, out + (out_size - 1),
                                              NPTS, KE, (long long)NPTS * D, CC);

    // ---------------- Decoder ----------------
    conv(Qb, dec_c1.w, dec_c1.b, G2, 8, 64, 64, 64, 128, 3, 3, 1, 1, ACT_NONE, nullptr); // x1
    hipMemcpyAsync(X2, G2, (size_t)4194304 * 4, hipMemcpyDeviceToDevice, stream);
    for (int r = 0; r < 2; ++r) resblock(X2, dec_res[r], EPS_RES, 128, 8, 64, 64);       // x2
    add_inplace_kernel<<<eblocks(4194304), 256, 0, stream>>>(G2, X2, 4194304);           // x1+x2

    // up[0]: conv -> BN -> shuffle -> PReLU   (8,128,64,64) -> (8,128,128,128)
    conv(G2, up[0].w, up[0].b, Ab, 8, 128, 64, 64, 512, 3, 3, 1, 1, ACT_NONE, nullptr);
    bn_stats_kernel<<<512, 256, 0, stream>>>(Ab, d_mean, d_var, 512, 8, 4096);
    bn_shuffle_prelu_kernel<<<eblocks(16777216), 256, 0, stream>>>(Ab, Bb, d_mean, d_var,
        up[0].g, up[0].bb, up[0].a, EPS_UP, 8, 128, 64, 64);

    // up[1]: (8,128,128,128) -> (8,128,256,256)
    conv(Bb, up[1].w, up[1].b, Cb, 8, 128, 128, 128, 512, 3, 3, 1, 1, ACT_NONE, nullptr);
    bn_stats_kernel<<<512, 256, 0, stream>>>(Cb, d_mean, d_var, 512, 8, 16384);
    bn_shuffle_prelu_kernel<<<eblocks(67108864), 256, 0, stream>>>(Cb, Db, d_mean, d_var,
        up[1].g, up[1].bb, up[1].a, EPS_UP, 8, 128, 128, 128);

    // dec_c2 (+PReLU) -> reuse Cb; dec_c3 -> x_recon straight into d_out[1..]
    conv(Db, dec_c2.w, dec_c2.b, Cb, 8, 128, 256, 256, 64, 3, 3, 1, 1, ACT_PRELU, a_out);
    conv(Cb, dec_c3.w, dec_c3.b, out + 1, 8, 64, 256, 256, 3, 3, 3, 1, 1, ACT_NONE, nullptr);
}
